// Model_5583457485578
// MI455X (gfx1250) — matrix operand order repaired and checked by reading, not compiled, not run on hardware
//
#include <hip/hip_runtime.h>
#include <math.h>

// ---------------------------------------------------------------------------
// GraphCodeBERT clone-detection forward for MI455X (gfx1250), wave32 + WMMA.
// All dense math runs through V_WMMA_F32_16X16X32_BF16 (bf16 in, f32 acc).
// Weight-transposed GEMMs stage their B panel into LDS with the Tensor Data
// Mover (tensor_load_to_lds, double-buffered, TENSORcnt-synced) with TDM LDS
// padding for bank-conflict-free fragment reads.
// Assumptions (compile-only environment, cannot verify at runtime):
//   * d_in[0..7] = ids1,pos1,mask1,dfg1, ids2,pos2,mask2,dfg2 (int32; bool->i32)
//   * d_in[8..]  = params flattened in sorted-key pytree order (see enum map)
//   * ws_size >= ~540 MB
// ---------------------------------------------------------------------------

#define B2   16
#define SEQL 512
#define HID  768
#define NLAY 6
#define CTXW 20
#define DMX  64
#define NDFG (B2*DMX)      // 1024
#define ROWS (B2*SEQL)     // 8192
#define FFH  3072
#define GRU3 2304
#define ALPHA 0.2f

#define KC   64            // K-chunk staged per TDM transfer
#define BROW 72            // LDS row stride in bf16 (32 dwords + 4 pad dwords)

typedef __bf16 bf16_t;
typedef __attribute__((ext_vector_type(16))) __bf16 bfrag_t;  // 32B WMMA frag
typedef __attribute__((ext_vector_type(8)))  __bf16 bhalf_t;  // 16B chunk
typedef __attribute__((ext_vector_type(8)))  float  f8_t;     // C/D frag
typedef __attribute__((ext_vector_type(4)))  unsigned int u32x4;
typedef __attribute__((ext_vector_type(8)))  int i32x8;
typedef __attribute__((ext_vector_type(4)))  int i32x4;

#define DEV __device__ __forceinline__

DEV bf16_t f2bf(float f) {
  unsigned u = __builtin_bit_cast(unsigned, f);
  u += 0x7fffu + ((u >> 16) & 1u);                 // round to nearest even
  unsigned short s = (unsigned short)(u >> 16);
  return __builtin_bit_cast(bf16_t, s);
}

// ---------------------------------------------------------------------------
// Fragment loaders (contiguous k order chosen here; A and B use the same one):
//   A 16x32 bf16: lane m+16g holds 16 contiguous K at [k+16g, +16)
//   B 32x16 bf16: lane n+16g holds 16 contiguous K at [k+16g, +16)
//   C 16x16 f32 : VGPR r -> M = r + 8g, N = lane & 15
// ---------------------------------------------------------------------------
DEV bfrag_t load_frag_A(const bf16_t* row, int kk, int g) {
  const bf16_t* p = row + kk + g * 16;
  union { bfrag_t v; bhalf_t h[2]; } u;
  u.h[0] = *(const bhalf_t*)(p);
  u.h[1] = *(const bhalf_t*)(p + 8);
  return u.v;
}
DEV bfrag_t load_frag_B_rowK(const bf16_t* row, int kk, int g) {
  union { bfrag_t v; bhalf_t h[2]; } u;
  const bf16_t* p = row + kk + g * 16;
  u.h[0] = *(const bhalf_t*)(p);
  u.h[1] = *(const bhalf_t*)(p + 8);
  return u.v;
}
DEV bfrag_t load_frag_B_colK(const bf16_t* base, long ldb, int col, int kk, int g) {
  bfrag_t v;
#pragma unroll
  for (int e = 0; e < 16; ++e)
    v[e] = base[(long)(kk + g * 16 + e) * ldb + col];
  return v;
}

// ---------------------------------------------------------------------------
// TDM: stage a 64-row x KC-col bf16 tile of B[N,K] (row-major, ldb elems)
// into LDS at lptr. D# per ISA ch.8: data_size=4B, 2D tile (32 dwords x 64),
// pad_enable with 4 dwords of pad every 32 dwords -> LDS row stride 36 dwords
// (BROW bf16), which makes the 16-lane fragment reads bank-conflict-free.
// ---------------------------------------------------------------------------
DEV void tdm_stage_b(const bf16_t* gptr, long ldb, const bf16_t* lptr) {
  unsigned long ga = (unsigned long)gptr;
  unsigned      la = (unsigned)(unsigned long)lptr;
  u32x4 g0;
  g0[0] = 1u;                                         // count=1, user D#
  g0[1] = la;                                         // lds_addr (bytes)
  g0[2] = (unsigned)(ga & 0xffffffffu);               // global_addr[31:0]
  g0[3] = (unsigned)((ga >> 32) & 0x01ffffffu) | (2u << 30);  // [56:32] | type=2

  const unsigned td0 = KC / 2;        // tensor_dim0 (dwords)
  const unsigned td1 = 64u;           // tensor_dim1 (rows)
  const unsigned tl0 = KC / 2;        // tile_dim0 (dwords)
  const unsigned tl1 = 64u;           // tile_dim1 (rows)
  unsigned long st0 = (unsigned long)(ldb / 2);       // dim0 stride (dwords)

  i32x8 g1;
  g1[0] = (int)((2u << 16) | (1u << 20) | (4u << 22) | (3u << 25));
  //            data_size=4B  pad_en      interval=32dw  amount=4dw
  g1[1] = (int)((td0 & 0xffffu) << 16);                         // tdim0[15:0]
  g1[2] = (int)(((td0 >> 16) & 0xffffu) | ((td1 & 0xffffu) << 16));
  g1[3] = (int)(((td1 >> 16) & 0xffffu) | ((tl0 & 0xffffu) << 16));
  g1[4] = (int)(tl1 & 0xffffu);                                 // tile_dim2=0
  g1[5] = (int)(unsigned)(st0 & 0xffffffffu);
  g1[6] = (int)(unsigned)((st0 >> 32) & 0xffffu);               // stride1=0
  g1[7] = 0;
  i32x4 z4 = {0, 0, 0, 0};
#if defined(__clang_major__) && (__clang_major__ >= 23)
  i32x8 z8 = {0, 0, 0, 0, 0, 0, 0, 0};
  __builtin_amdgcn_tensor_load_to_lds(g0, g1, z4, z4, z8, 0);
#else
  __builtin_amdgcn_tensor_load_to_lds(g0, g1, z4, z4, 0);
#endif
}

// ---------------------------------------------------------------------------
// GEMM (weightsT): C[M,N] = A[M,K] * B[N,K]^T + bias. bf16 in, f32 acc.
// 128 threads / 4 waves; 64x64 block tile; 32x32 per wave (2x2 WMMA tiles).
// B panel TDM-staged into LDS, double buffered over KC chunks; wave0 issues,
// waits TENSORcnt, workgroup barrier hands the tile to all waves.
// ---------------------------------------------------------------------------
__global__ void __launch_bounds__(128)
gemm_bf16_wmma_wT(const bf16_t* A, long lda, long sA,
                  const bf16_t* B, long ldb, long sB,
                  float* Cf, bf16_t* Cb, long ldc, long sC,
                  const float* bias, int M, int N, int K) {
  __shared__ __align__(16) bf16_t bsm[2][64 * BROW];

  int batch = blockIdx.z;
  A += (long)batch * sA;
  B += (long)batch * sB;
  if (Cf) Cf += (long)batch * sC;
  if (Cb) Cb += (long)batch * sC;

  const int lane = threadIdx.x & 31;
  const int wave = threadIdx.x >> 5;
  const int g    = lane >> 4;
  const int mn   = lane & 15;
  const int m0   = blockIdx.y * 64 + (wave >> 1) * 32;
  const int nl   = (wave & 1) * 32;            // local n within block tile
  const int n0   = blockIdx.x * 64 + nl;

  const bf16_t* Bblk = B + (long)(blockIdx.x * 64) * ldb;
  const bool issuer = (threadIdx.x < 32);
  const int nch = K / KC;

  if (issuer) tdm_stage_b(Bblk, ldb, &bsm[0][0]);

  f8_t acc[2][2] = {};
  for (int c = 0; c < nch; ++c) {
    if (issuer) {
      if (c + 1 < nch) {
        tdm_stage_b(Bblk + (long)(c + 1) * KC, ldb, &bsm[(c + 1) & 1][0]);
        __builtin_amdgcn_s_wait_tensorcnt(1);  // chunk c landed
      } else {
        __builtin_amdgcn_s_wait_tensorcnt(0);
      }
    }
    __syncthreads();
    const bf16_t* bs = &bsm[c & 1][0];
#pragma unroll
    for (int ks = 0; ks < KC; ks += 32) {
      const int kk = c * KC + ks;
      bfrag_t a0 = load_frag_A(A + (long)(m0 + mn) * lda, kk, g);
      bfrag_t a1 = load_frag_A(A + (long)(m0 + 16 + mn) * lda, kk, g);
      bfrag_t b0 = load_frag_B_rowK(bs + (nl + mn) * BROW, ks, g);
      bfrag_t b1 = load_frag_B_rowK(bs + (nl + 16 + mn) * BROW, ks, g);
      acc[0][0] = __builtin_amdgcn_wmma_f32_16x16x32_bf16(false, a0, false, b0, (short)0, acc[0][0], false, false);
      acc[0][1] = __builtin_amdgcn_wmma_f32_16x16x32_bf16(false, a0, false, b1, (short)0, acc[0][1], false, false);
      acc[1][0] = __builtin_amdgcn_wmma_f32_16x16x32_bf16(false, a1, false, b0, (short)0, acc[1][0], false, false);
      acc[1][1] = __builtin_amdgcn_wmma_f32_16x16x32_bf16(false, a1, false, b1, (short)0, acc[1][1], false, false);
    }
    __syncthreads();
  }
#pragma unroll
  for (int i = 0; i < 2; ++i)
#pragma unroll
    for (int j = 0; j < 2; ++j) {
      int col = blockIdx.x * 64 + nl + j * 16 + mn;
      float bv = bias ? bias[col] : 0.0f;
      int rowb = m0 + i * 16 + 8 * g;
#pragma unroll
      for (int r = 0; r < 8; ++r) {
        float v = acc[i][j][r] + bv;
        long off = (long)(rowb + r) * ldc + col;
        if (Cf) Cf[off] = v;
        if (Cb) Cb[off] = f2bf(v);
      }
    }
  (void)n0;
}

// GEMM (non-transposed B [K,N], the P@V path): direct global loads.
__global__ void __launch_bounds__(128)
gemm_bf16_wmma_nn(const bf16_t* A, long lda, long sA,
                  const bf16_t* B, long ldb, long sB,
                  float* Cf, bf16_t* Cb, long ldc, long sC,
                  const float* bias, int M, int N, int K) {
  int batch = blockIdx.z;
  A += (long)batch * sA;
  B += (long)batch * sB;
  if (Cf) Cf += (long)batch * sC;
  if (Cb) Cb += (long)batch * sC;

  const int lane = threadIdx.x & 31;
  const int wave = threadIdx.x >> 5;
  const int g    = lane >> 4;
  const int mn   = lane & 15;
  const int m0   = blockIdx.y * 64 + (wave >> 1) * 32;
  const int n0   = blockIdx.x * 64 + (wave & 1) * 32;

  f8_t acc[2][2] = {};
  for (int kk = 0; kk < K; kk += 32) {
    bfrag_t a0 = load_frag_A(A + (long)(m0 + mn) * lda, kk, g);
    bfrag_t a1 = load_frag_A(A + (long)(m0 + 16 + mn) * lda, kk, g);
    bfrag_t b0 = load_frag_B_colK(B, ldb, n0 + mn, kk, g);
    bfrag_t b1 = load_frag_B_colK(B, ldb, n0 + 16 + mn, kk, g);
    acc[0][0] = __builtin_amdgcn_wmma_f32_16x16x32_bf16(false, a0, false, b0, (short)0, acc[0][0], false, false);
    acc[0][1] = __builtin_amdgcn_wmma_f32_16x16x32_bf16(false, a0, false, b1, (short)0, acc[0][1], false, false);
    acc[1][0] = __builtin_amdgcn_wmma_f32_16x16x32_bf16(false, a1, false, b0, (short)0, acc[1][0], false, false);
    acc[1][1] = __builtin_amdgcn_wmma_f32_16x16x32_bf16(false, a1, false, b1, (short)0, acc[1][1], false, false);
  }
#pragma unroll
  for (int i = 0; i < 2; ++i)
#pragma unroll
    for (int j = 0; j < 2; ++j) {
      int col = n0 + j * 16 + mn;
      float bv = bias ? bias[col] : 0.0f;
      int rowb = m0 + i * 16 + 8 * g;
#pragma unroll
      for (int r = 0; r < 8; ++r) {
        float v = acc[i][j][r] + bv;
        long off = (long)(rowb + r) * ldc + col;
        if (Cf) Cf[off] = v;
        if (Cb) Cb[off] = f2bf(v);
      }
    }
}

// ---------------------------------------------------------------------------
// Elementwise / reduction helper kernels
// ---------------------------------------------------------------------------
__global__ void cvt_f32_bf16(const float* x, bf16_t* y, long n) {
  long st = (long)gridDim.x * blockDim.x;
  for (long i = (long)blockIdx.x * blockDim.x + threadIdx.x; i < n; i += st)
    y[i] = f2bf(x[i]);
}

__global__ void init_h_zero(float* h, bf16_t* hb, long n) {
  long st = (long)gridDim.x * blockDim.x;
  for (long i = (long)blockIdx.x * blockDim.x + threadIdx.x; i < n; i += st) {
    h[i] = 0.0f;
    hb[i] = f2bf(0.0f);
  }
}

__global__ void interleave_i32(const int* a, const int* b, int* out, long perSeq) {
  long total = perSeq * B2;
  long st = (long)gridDim.x * blockDim.x;
  for (long i = (long)blockIdx.x * blockDim.x + threadIdx.x; i < total; i += st) {
    long s = i / perSeq, j = i % perSeq;
    const int* src = (s & 1) ? b : a;
    out[i] = src[(s >> 1) * perSeq + j];
  }
}

__global__ void seq_stats(const int* pos, int* dfg_index, int* dfg_len) {
  int s = blockIdx.x;
  __shared__ int r1[256], r2[256];
  int c1 = 0, c2 = 0;
  for (int j = threadIdx.x; j < SEQL; j += 256) {
    int p = pos[s * SEQL + j];
    c1 += (p >= 2);
    c2 += (p == 0);
  }
  r1[threadIdx.x] = c1; r2[threadIdx.x] = c2;
  __syncthreads();
  for (int o = 128; o > 0; o >>= 1) {
    if (threadIdx.x < o) { r1[threadIdx.x] += r1[threadIdx.x + o]; r2[threadIdx.x] += r2[threadIdx.x + o]; }
    __syncthreads();
  }
  if (threadIdx.x == 0) { dfg_index[s] = r1[0]; dfg_len[s] = r2[0]; }
}

__global__ void gather_dfg(const int* d1, const int* d2, const int* dfg_index,
                           int* flat_ids, int* lens) {
  int n = blockIdx.x * blockDim.x + threadIdx.x;
  if (n >= NDFG) return;
  int s = n / DMX, dd = n % DMX;
  // dfg was concatenated (NOT interleaved) along batch dim in the reference
  const int* dsrc = (s < 8) ? (d1 + (long)s * SEQL * CTXW)
                            : (d2 + (long)(s - 8) * SEQL * CTXW);
  int idx = dfg_index[s] + dd;
  int cnt = 0;
  for (int c = 0; c < CTXW; ++c) {
    int id = (idx < SEQL) ? dsrc[(long)idx * CTXW + c] : 1;
    flat_ids[(long)n * CTXW + c] = id;
    cnt += (id != 1);
  }
  lens[n] = cnt > 0 ? cnt : 1;
}

__global__ void gather_rows_bf16(const int* ids, const float* table, bf16_t* out) {
  long row = blockIdx.x;
  const float* src = table + (long)ids[row] * HID;
  bf16_t* dst = out + row * HID;
  for (int k = threadIdx.x; k < HID; k += blockDim.x) dst[k] = f2bf(src[k]);
}

__global__ void gather_rows_f32(const int* ids, const float* table, float* out) {
  long row = blockIdx.x;
  const float* src = table + (long)ids[row] * HID;
  float* dst = out + row * HID;
  for (int k = threadIdx.x; k < HID; k += blockDim.x) dst[k] = src[k];
}

__global__ void gru_step(const float* gi, const float* gh, float* h, bf16_t* hb,
                         const int* lens, int t) {
  long idx = (long)blockIdx.x * blockDim.x + threadIdx.x;
  if (idx >= (long)NDFG * HID) return;
  int n = (int)(idx / HID);
  int k = (int)(idx % HID);
  float hv = h[idx];
  if (t < lens[n]) {
    const float* gir = gi + (long)n * GRU3;
    const float* ghr = gh + (long)n * GRU3;
    float r = 1.0f / (1.0f + expf(-(gir[k] + ghr[k])));
    float z = 1.0f / (1.0f + expf(-(gir[HID + k] + ghr[HID + k])));
    float nn = tanhf(gir[2 * HID + k] + r * ghr[2 * HID + k]);
    hv = (1.0f - z) * nn + z * hv;
  }
  h[idx] = hv;
  hb[idx] = f2bf(hv);
}

__global__ void reduce_max_int(const int* v, int n, int* out) {
  __shared__ int red[256];
  int m = 0;
  for (int i = threadIdx.x; i < n; i += 256) m = v[i] > m ? v[i] : m;
  red[threadIdx.x] = m;
  __syncthreads();
  for (int o = 128; o > 0; o >>= 1) {
    if (threadIdx.x < o) red[threadIdx.x] = red[threadIdx.x] > red[threadIdx.x + o] ? red[threadIdx.x] : red[threadIdx.x + o];
    __syncthreads();
  }
  if (threadIdx.x == 0) out[0] = red[0];
}

__global__ void make_dfg_out(const float* h, const int* lens, const int* maxlen,
                             const float* pos_emb, float* dfg_out) {
  long n = blockIdx.x;                     // 0..1023
  int dd = (int)(n % DMX);
  bool keep = (lens[n] == maxlen[0]);
  const float* hr = h + n * HID;
  const float* pe = pos_emb + (long)dd * HID;
  float* o = dfg_out + n * HID;
  for (int k = threadIdx.x; k < HID; k += blockDim.x)
    o[k] = (keep ? hr[k] : 0.0f) + pe[k];
}

__global__ void node_avg(const float* emb, const int* pos, const int* m1, const int* m2,
                         float* avg) {
  int s = blockIdx.y, i = blockIdx.x;
  const int* prow = pos + s * SEQL;
  float* out = avg + ((long)s * SEQL + i) * HID;
  if (prow[i] != 0) {
    for (int k = threadIdx.x; k < HID; k += blockDim.x) out[k] = 0.0f;
    return;
  }
  const int* mb = (s & 1) ? m2 : m1;
  const int* mrow = mb + ((long)(s >> 1) * SEQL + i) * SEQL;
  float acc0 = 0.f, acc1 = 0.f, acc2 = 0.f;
  int cnt = 0;
  for (int j = 0; j < SEQL; ++j) {
    if (prow[j] >= 2 && mrow[j] != 0) {
      const float* er = emb + ((long)s * SEQL + j) * HID;
      acc0 += er[threadIdx.x];
      acc1 += er[threadIdx.x + 256];
      acc2 += er[threadIdx.x + 512];
      cnt++;
    }
  }
  float inv = 1.0f / ((float)cnt + 1e-10f);
  out[threadIdx.x] = acc0 * inv;
  out[threadIdx.x + 256] = acc1 * inv;
  out[threadIdx.x + 512] = acc2 * inv;
}

__global__ void blend_dfg(float* avg, const float* dfg_out,
                          const int* dfg_index, const int* dfg_len) {
  int s = blockIdx.y, i = blockIdx.x;
  int slot = i - dfg_index[s];
  if (slot < 0 || slot >= dfg_len[s]) return;
  int cs = slot < (DMX - 1) ? slot : (DMX - 1);
  float* a = avg + ((long)s * SEQL + i) * HID;
  const float* d = dfg_out + ((long)s * DMX + cs) * HID;
  for (int k = threadIdx.x; k < HID; k += blockDim.x)
    a[k] = (1.0f - ALPHA) * a[k] + ALPHA * d[k];
}

__global__ void softmax_row_bf16(const float* S, bf16_t* P, const int* m1, const int* m2,
                                 float scale, int useMask) {
  long row = blockIdx.x;                 // 0..B2*SEQL-1
  int s = (int)(row / SEQL), i = (int)(row % SEQL);
  const float* sr = S + row * SEQL;
  bf16_t* pr = P + row * SEQL;
  const int* mrow = nullptr;
  if (useMask) {
    const int* mb = (s & 1) ? m2 : m1;
    mrow = mb + ((long)(s >> 1) * SEQL + i) * SEQL;
  }
  __shared__ float red[256];
  float v[2];
  float lmax = -3.0e38f;
#pragma unroll
  for (int u = 0; u < 2; ++u) {
    int j = threadIdx.x + u * 256;
    float x = sr[j] * scale;
    if (useMask && mrow[j] == 0) x += -1e9f;
    v[u] = x;
    lmax = fmaxf(lmax, x);
  }
  red[threadIdx.x] = lmax;
  __syncthreads();
  for (int o = 128; o > 0; o >>= 1) {
    if (threadIdx.x < o) red[threadIdx.x] = fmaxf(red[threadIdx.x], red[threadIdx.x + o]);
    __syncthreads();
  }
  float m = red[0];
  __syncthreads();
  float ls = 0.f;
#pragma unroll
  for (int u = 0; u < 2; ++u) { v[u] = expf(v[u] - m); ls += v[u]; }
  red[threadIdx.x] = ls;
  __syncthreads();
  for (int o = 128; o > 0; o >>= 1) {
    if (threadIdx.x < o) red[threadIdx.x] += red[threadIdx.x + o];
    __syncthreads();
  }
  float inv = 1.0f / red[0];
#pragma unroll
  for (int u = 0; u < 2; ++u) pr[threadIdx.x + u * 256] = f2bf(v[u] * inv);
}

__global__ void mix_embed(const float* inp, const float* avg, const int* pos,
                          const float* pos_emb, const float* type_emb, float* out) {
  long row = blockIdx.x;
  int p = pos[row];
  const float* src = (p == 0) ? (avg + row * HID) : (inp + row * HID);
  const float* pe = pos_emb + (long)p * HID;
  float* o = out + row * HID;
  for (int k = threadIdx.x; k < HID; k += blockDim.x)
    o[k] = src[k] + pe[k] + type_emb[k];
}

// LayerNorm over H=768, optional residual add (in-place on x).
__global__ void add_ln(float* x, const float* res, const float* g, const float* b) {
  long row = blockIdx.x;
  float* xr = x + row * HID;
  const float* rr = res ? res + row * HID : nullptr;
  __shared__ float red[256];
  float v[3];
#pragma unroll
  for (int u = 0; u < 3; ++u) {
    int k = threadIdx.x + u * 256;
    float t = xr[k];
    if (rr) t += rr[k];
    v[u] = t;
  }
  red[threadIdx.x] = v[0] + v[1] + v[2];
  __syncthreads();
  for (int o = 128; o > 0; o >>= 1) {
    if (threadIdx.x < o) red[threadIdx.x] += red[threadIdx.x + o];
    __syncthreads();
  }
  float mean = red[0] / (float)HID;
  __syncthreads();
  float lv = 0.f;
#pragma unroll
  for (int u = 0; u < 3; ++u) { float d = v[u] - mean; lv += d * d; }
  red[threadIdx.x] = lv;
  __syncthreads();
  for (int o = 128; o > 0; o >>= 1) {
    if (threadIdx.x < o) red[threadIdx.x] += red[threadIdx.x + o];
    __syncthreads();
  }
  float rstd = rsqrtf(red[0] / (float)HID + 1e-12f);
#pragma unroll
  for (int u = 0; u < 3; ++u) {
    int k = threadIdx.x + u * 256;
    xr[k] = (v[u] - mean) * rstd * g[k] + b[k];
  }
}

__global__ void gelu_bf16(const float* x, bf16_t* y, long n) {
  long st = (long)gridDim.x * blockDim.x;
  for (long i = (long)blockIdx.x * blockDim.x + threadIdx.x; i < n; i += st) {
    float v = x[i];
    y[i] = f2bf(0.5f * v * (1.0f + erff(v * 0.70710678118654752f)));
  }
}

__global__ void classifier_head(const float* h, const float* dw, const float* db,
                                const float* ow, const float* ob, float* out) {
  int b = blockIdx.x;                       // 0..7 pairs
  __shared__ float xs[HID];
  __shared__ float red[256];
  const float* c0 = h + (long)(2 * b) * SEQL * HID;     // CLS of seq 2b
  const float* c1 = c0 + (long)SEQL * HID;              // CLS of seq 2b+1
  for (int o = threadIdx.x; o < HID; o += 256) {
    const float* w = dw + (long)o * (2 * HID);
    float acc = db[o];
    for (int k = 0; k < HID; ++k) acc += c0[k] * w[k];
    for (int k = 0; k < HID; ++k) acc += c1[k] * w[HID + k];
    xs[o] = tanhf(acc);
  }
  __syncthreads();
  float l0 = 0.f, l1 = 0.f;
  for (int o = threadIdx.x; o < HID; o += 256) {
    l0 += xs[o] * ow[o];
    l1 += xs[o] * ow[HID + o];
  }
  red[threadIdx.x] = l0;
  __syncthreads();
  for (int o = 128; o > 0; o >>= 1) { if (threadIdx.x < o) red[threadIdx.x] += red[threadIdx.x + o]; __syncthreads(); }
  float L0 = red[0];
  __syncthreads();
  red[threadIdx.x] = l1;
  __syncthreads();
  for (int o = 128; o > 0; o >>= 1) { if (threadIdx.x < o) red[threadIdx.x] += red[threadIdx.x + o]; __syncthreads(); }
  float L1 = red[0];
  if (threadIdx.x == 0) {
    L0 += ob[0]; L1 += ob[1];
    float m = fmaxf(L0, L1);
    float e0 = expf(L0 - m), e1 = expf(L1 - m);
    float s = e0 + e1;
    out[2 * b] = e0 / s;
    out[2 * b + 1] = e1 / s;
  }
}

// ---------------------------------------------------------------------------
// Host orchestration
// ---------------------------------------------------------------------------
static inline unsigned gsz(long n, long maxb = 16384) {
  long b = (n + 255) / 256;
  if (b > maxb) b = maxb;
  return (unsigned)(b > 0 ? b : 1);
}

static void run_gemm(hipStream_t st, const bf16_t* A, long lda, long sA,
                     const bf16_t* B, long ldb, long sB,
                     float* Cf, bf16_t* Cb, long ldc, long sC,
                     const float* bias, int M, int N, int K, int batch, bool bt) {
  dim3 grid((unsigned)(N / 64), (unsigned)(M / 64), (unsigned)batch);
  dim3 block(128);
  if (bt)
    gemm_bf16_wmma_wT<<<grid, block, 0, st>>>(A, lda, sA, B, ldb, sB, Cf, Cb, ldc, sC, bias, M, N, K);
  else
    gemm_bf16_wmma_nn<<<grid, block, 0, st>>>(A, lda, sA, B, ldb, sB, Cf, Cb, ldc, sC, bias, M, N, K);
}

extern "C" void kernel_launch(void* const* d_in, const int* in_sizes, int n_in,
                              void* d_out, int out_size, void* d_ws, size_t ws_size,
                              hipStream_t stream) {
  (void)in_sizes; (void)n_in; (void)out_size; (void)ws_size;

  auto pf = [&](int i) { return (const float*)d_in[i]; };
  auto pi = [&](int i) { return (const int*)d_in[i]; };

  const int *ids1 = pi(0), *posx1 = pi(1), *mask1 = pi(2), *dfg1 = pi(3);
  const int *ids2 = pi(4), *posx2 = pi(5), *mask2 = pi(6), *dfg2 = pi(7);

  // params (sorted pytree order)
  const float* cls_dense_b = pf(8);
  const float* cls_dense_w = pf(9);
  const float* cls_out_b   = pf(10);
  const float* cls_out_w   = pf(11);
  const float* emb_ln_b    = pf(12);
  const float* emb_ln_g    = pf(13);
  const float* ff_b        = pf(14);
  const float* ff_w        = pf(15);
  const float* gru_bh      = pf(16);
  const float* gru_bi      = pf(17);
  const float* gru_wh      = pf(18);
  const float* gru_wi      = pf(19);
  const float* k_b         = pf(20);
  const float* k_w         = pf(21);
  enum { F_FB1 = 0, F_FB2, F_FW1, F_FW2, F_KB, F_KW, F_LN1B, F_LN1G,
         F_LN2B, F_LN2G, F_OB, F_OW, F_QB, F_QW, F_VB, F_VW };
  auto LPf = [&](int l, int f) { return pf(22 + 16 * l + f); };
  const float* pos_emb  = pf(118);
  const float* q_b      = pf(119);
  const float* q_w      = pf(120);
  const float* type_emb = pf(121);
  const float* v_b      = pf(122);
  const float* v_w      = pf(123);
  const float* word_emb = pf(124);

  // ---- workspace bump allocator ----
  char* wsp = (char*)d_ws;
  size_t off = 0;
  auto alloc = [&](size_t bytes) -> void* {
    size_t a = (off + 255) & ~(size_t)255;
    void* p = wsp + a;
    off = a + bytes;
    return p;
  };

  int* ids_i     = (int*)alloc((size_t)ROWS * 4);
  int* pos_i     = (int*)alloc((size_t)ROWS * 4);
  int* dfg_index = (int*)alloc(B2 * 4);
  int* dfg_len   = (int*)alloc(B2 * 4);
  int* flat_ids  = (int*)alloc((size_t)NDFG * CTXW * 4);
  int* lens      = (int*)alloc(NDFG * 4);
  int* maxlen    = (int*)alloc(4);

  auto wcvt = [&](const float* src, long n) -> bf16_t* {
    bf16_t* dst = (bf16_t*)alloc((size_t)n * 2);
    cvt_f32_bf16<<<gsz(n), 256, 0, stream>>>(src, dst, n);
    return dst;
  };

  // bf16 weights
  bf16_t* wib  = wcvt(gru_wi, (long)GRU3 * HID);
  bf16_t* whb  = wcvt(gru_wh, (long)GRU3 * HID);
  bf16_t* qwb0 = wcvt(q_w, (long)HID * HID);
  bf16_t* kwb0 = wcvt(k_w, (long)HID * HID);
  bf16_t* vwb0 = wcvt(v_w, (long)HID * HID);
  bf16_t* ffb0 = wcvt(ff_w, (long)HID * HID);
  bf16_t *Lqw[NLAY], *Lkw[NLAY], *Lvw[NLAY], *Low[NLAY], *Lf1[NLAY], *Lf2[NLAY];
  for (int l = 0; l < NLAY; ++l) {
    Lqw[l] = wcvt(LPf(l, F_QW), (long)HID * HID);
    Lkw[l] = wcvt(LPf(l, F_KW), (long)HID * HID);
    Lvw[l] = wcvt(LPf(l, F_VW), (long)HID * HID);
    Low[l] = wcvt(LPf(l, F_OW), (long)HID * HID);
    Lf1[l] = wcvt(LPf(l, F_FW1), (long)FFH * HID);
    Lf2[l] = wcvt(LPf(l, F_FW2), (long)HID * FFH);
  }

  // activations
  bf16_t* xgru_b = (bf16_t*)alloc((size_t)NDFG * CTXW * HID * 2);
  float*  gi     = (float*)alloc((size_t)NDFG * GRU3 * 4);
  float*  gh     = (float*)alloc((size_t)NDFG * GRU3 * 4);
  float*  hbuf   = (float*)alloc((size_t)NDFG * HID * 4);
  bf16_t* hb     = (bf16_t*)alloc((size_t)NDFG * HID * 2);
  float*  dfgo   = (float*)alloc((size_t)B2 * DMX * HID * 4);
  float*  inp    = (float*)alloc((size_t)ROWS * HID * 4);
  float*  avg    = (float*)alloc((size_t)ROWS * HID * 4);
  bf16_t* xb     = (bf16_t*)alloc((size_t)ROWS * HID * 2);
  bf16_t* Qb     = (bf16_t*)alloc((size_t)ROWS * HID * 2);
  bf16_t* Kb     = (bf16_t*)alloc((size_t)ROWS * HID * 2);
  bf16_t* Vb     = (bf16_t*)alloc((size_t)ROWS * HID * 2);
  float*  scores = (float*)alloc((size_t)B2 * SEQL * SEQL * 4);
  bf16_t* Pb     = (bf16_t*)alloc((size_t)B2 * SEQL * SEQL * 2);
  float*  ctx    = (float*)alloc((size_t)ROWS * HID * 4);
  float*  attnf  = (float*)alloc((size_t)ROWS * HID * 4);
  float*  henc   = (float*)alloc((size_t)ROWS * HID * 4);
  float*  f1f    = (float*)alloc((size_t)ROWS * FFH * 4);
  bf16_t* f1b    = (bf16_t*)alloc((size_t)ROWS * FFH * 2);

  // ---- 1: interleave ids/pos ----
  interleave_i32<<<gsz(ROWS), 256, 0, stream>>>(ids1, ids2, ids_i, SEQL);
  interleave_i32<<<gsz(ROWS), 256, 0, stream>>>(posx1, posx2, pos_i, SEQL);
  seq_stats<<<B2, 256, 0, stream>>>(pos_i, dfg_index, dfg_len);

  // ---- 2: DFG context gather + GRU ----
  gather_dfg<<<(NDFG + 255) / 256, 256, 0, stream>>>(dfg1, dfg2, dfg_index, flat_ids, lens);
  gather_rows_bf16<<<NDFG * CTXW, 256, 0, stream>>>(flat_ids, word_emb, xgru_b);
  init_h_zero<<<gsz((long)NDFG * HID), 256, 0, stream>>>(hbuf, hb, (long)NDFG * HID);
  for (int t = 0; t < CTXW; ++t) {
    run_gemm(stream, xgru_b + (long)t * HID, (long)CTXW * HID, 0,
             wib, HID, 0, gi, nullptr, GRU3, 0, gru_bi, NDFG, GRU3, HID, 1, true);
    run_gemm(stream, hb, HID, 0, whb, HID, 0, gh, nullptr, GRU3, 0, gru_bh,
             NDFG, GRU3, HID, 1, true);
    gru_step<<<(unsigned)((long)NDFG * HID / 256), 256, 0, stream>>>(gi, gh, hbuf, hb, lens, t);
  }
  reduce_max_int<<<1, 256, 0, stream>>>(lens, NDFG, maxlen);
  make_dfg_out<<<NDFG, 256, 0, stream>>>(hbuf, lens, maxlen, pos_emb, dfgo);

  // ---- 3: token embeddings + graph-average mixer ----
  gather_rows_f32<<<ROWS, 256, 0, stream>>>(ids_i, word_emb, inp);
  node_avg<<<dim3(SEQL, B2), 256, 0, stream>>>(inp, pos_i, mask1, mask2, avg);
  blend_dfg<<<dim3(SEQL, B2), 256, 0, stream>>>(avg, dfgo, dfg_index, dfg_len);

  // ---- 4: pre-mixer MHA (4 heads, d=192, no mask) ----
  cvt_f32_bf16<<<gsz((long)ROWS * HID), 256, 0, stream>>>(avg, xb, (long)ROWS * HID);
  run_gemm(stream, xb, HID, 0, qwb0, HID, 0, nullptr, Qb, HID, 0, q_b, ROWS, HID, HID, 1, true);
  run_gemm(stream, xb, HID, 0, kwb0, HID, 0, nullptr, Kb, HID, 0, k_b, ROWS, HID, HID, 1, true);
  run_gemm(stream, xb, HID, 0, vwb0, HID, 0, nullptr, Vb, HID, 0, v_b, ROWS, HID, HID, 1, true);
  for (int h = 0; h < 4; ++h) {
    const int dh = 192;
    run_gemm(stream, Qb + h * dh, HID, (long)SEQL * HID, Kb + h * dh, HID, (long)SEQL * HID,
             scores, nullptr, SEQL, (long)SEQL * SEQL, nullptr, SEQL, SEQL, dh, B2, true);
    softmax_row_bf16<<<ROWS, 256, 0, stream>>>(scores, Pb, nullptr, nullptr,
                                               0.072168783648703220563f /*1/sqrt(192)*/, 0);
    run_gemm(stream, Pb, SEQL, (long)SEQL * SEQL, Vb + h * dh, HID, (long)SEQL * HID,
             ctx + h * dh, nullptr, HID, (long)SEQL * HID, nullptr, SEQL, dh, SEQL, B2, false);
  }
  cvt_f32_bf16<<<gsz((long)ROWS * HID), 256, 0, stream>>>(ctx, xb, (long)ROWS * HID);
  run_gemm(stream, xb, HID, 0, ffb0, HID, 0, avg, nullptr, HID, 0, ff_b, ROWS, HID, HID, 1, true);

  // ---- 5: RoBERTa embeddings ----
  mix_embed<<<ROWS, 256, 0, stream>>>(inp, avg, pos_i, pos_emb, type_emb, henc);
  add_ln<<<ROWS, 256, 0, stream>>>(henc, nullptr, emb_ln_g, emb_ln_b);

  // ---- 6: encoder (6 layers, 12 heads, d=64, masked) ----
  for (int l = 0; l < NLAY; ++l) {
    cvt_f32_bf16<<<gsz((long)ROWS * HID), 256, 0, stream>>>(henc, xb, (long)ROWS * HID);
    run_gemm(stream, xb, HID, 0, Lqw[l], HID, 0, nullptr, Qb, HID, 0, LPf(l, F_QB), ROWS, HID, HID, 1, true);
    run_gemm(stream, xb, HID, 0, Lkw[l], HID, 0, nullptr, Kb, HID, 0, LPf(l, F_KB), ROWS, HID, HID, 1, true);
    run_gemm(stream, xb, HID, 0, Lvw[l], HID, 0, nullptr, Vb, HID, 0, LPf(l, F_VB), ROWS, HID, HID, 1, true);
    for (int h = 0; h < 12; ++h) {
      const int dh = 64;
      run_gemm(stream, Qb + h * dh, HID, (long)SEQL * HID, Kb + h * dh, HID, (long)SEQL * HID,
               scores, nullptr, SEQL, (long)SEQL * SEQL, nullptr, SEQL, SEQL, dh, B2, true);
      softmax_row_bf16<<<ROWS, 256, 0, stream>>>(scores, Pb, mask1, mask2, 0.125f, 1);
      run_gemm(stream, Pb, SEQL, (long)SEQL * SEQL, Vb + h * dh, HID, (long)SEQL * HID,
               ctx + h * dh, nullptr, HID, (long)SEQL * HID, nullptr, SEQL, dh, SEQL, B2, false);
    }
    cvt_f32_bf16<<<gsz((long)ROWS * HID), 256, 0, stream>>>(ctx, xb, (long)ROWS * HID);
    run_gemm(stream, xb, HID, 0, Low[l], HID, 0, attnf, nullptr, HID, 0, LPf(l, F_OB), ROWS, HID, HID, 1, true);
    add_ln<<<ROWS, 256, 0, stream>>>(henc, attnf, LPf(l, F_LN1G), LPf(l, F_LN1B));
    cvt_f32_bf16<<<gsz((long)ROWS * HID), 256, 0, stream>>>(henc, xb, (long)ROWS * HID);
    run_gemm(stream, xb, HID, 0, Lf1[l], HID, 0, f1f, nullptr, FFH, 0, LPf(l, F_FB1), ROWS, FFH, HID, 1, true);
    gelu_bf16<<<gsz((long)ROWS * FFH), 256, 0, stream>>>(f1f, f1b, (long)ROWS * FFH);
    run_gemm(stream, f1b, FFH, 0, Lf2[l], FFH, 0, attnf, nullptr, HID, 0, LPf(l, F_FB2), ROWS, HID, FFH, 1, true);
    add_ln<<<ROWS, 256, 0, stream>>>(henc, attnf, LPf(l, F_LN2G), LPf(l, F_LN2B));
  }

  // ---- 7: paired-CLS classifier + softmax ----
  classifier_head<<<8, 256, 0, stream>>>(henc, cls_dense_w, cls_dense_b,
                                         cls_out_w, cls_out_b, (float*)d_out);
}
